// HungarianMatcher_77704548319892
// MI455X (gfx1250) — matrix operand order repaired and checked by reading, not compiled, not run on hardware
//
#include <hip/hip_runtime.h>
#include <math.h>

#define BS 2
#define Qn 900
#define Cn 91
#define Tn 100
#define Hn 128
#define Wn 128
#define Pn 1024
#define MT ((Qn + 15) / 16)   // 57 M tiles
#define NT ((Tn + 15) / 16)   // 7  N tiles

typedef __attribute__((ext_vector_type(16))) _Float16 v16h;
typedef __attribute__((ext_vector_type(8)))  _Float16 v8h;
typedef __attribute__((ext_vector_type(8)))  float    v8f;

__device__ __forceinline__ float softplus_f(float x) {
    return (x > 0.f) ? (x + log1pf(expf(-x))) : log1pf(expf(x));
}
__device__ __forceinline__ float sigmoid_f(float x) {
    return 1.f / (1.f + expf(-x));
}
__device__ __forceinline__ float samp2d(const float* __restrict__ m, int yi, int xi) {
    bool v = (xi >= 0) && (xi < Wn) && (yi >= 0) && (yi < Hn);
    int xc = min(max(xi, 0), Wn - 1);
    int yc = min(max(yi, 0), Hn - 1);
    return v ? m[yc * Wn + xc] : 0.f;
}

// ---------------- Kernel 1: nearest-sample tgt masks -> f16 tm rows + row sums
__global__ void hm_tm_sample(const int* __restrict__ tmasks,
                             const float* __restrict__ pts,
                             _Float16* __restrict__ tmh,
                             float* __restrict__ tmsum) {
    int b = blockIdx.x / Tn;
    int t = blockIdx.x % Tn;
    const int* mask = tmasks + ((size_t)(b * Tn + t)) * (Hn * Wn);
    const float* pc = pts + (size_t)b * Pn * 2;
    float s = 0.f;
    for (int p = threadIdx.x; p < Pn; p += blockDim.x) {
        float x = pc[2 * p + 0] * Wn - 0.5f;
        float y = pc[2 * p + 1] * Hn - 0.5f;
        int xi = (int)rintf(x);   // round half-to-even, matches jnp.round
        int yi = (int)rintf(y);
        bool valid = (xi >= 0) && (xi < Wn) && (yi >= 0) && (yi < Hn);
        int xc = min(max(xi, 0), Wn - 1);
        int yc = min(max(yi, 0), Hn - 1);
        float v = valid ? (float)mask[yc * Wn + xc] : 0.f;
        tmh[((size_t)(b * Tn + t)) * Pn + p] = (_Float16)v;
        s += v;
    }
    __shared__ float red[256];
    red[threadIdx.x] = s;
    __syncthreads();
    for (int o = blockDim.x / 2; o > 0; o >>= 1) {
        if (threadIdx.x < (unsigned)o) red[threadIdx.x] += red[threadIdx.x + o];
        __syncthreads();
    }
    if (threadIdx.x == 0) tmsum[b * Tn + t] = red[0];
}

// ---------------- Kernel 2: bilinear-sample pred masks -> pm/sm f16 + row sums
__global__ void hm_pm_sample(const float* __restrict__ pmasks,
                             const float* __restrict__ pts,
                             _Float16* __restrict__ pmh,
                             _Float16* __restrict__ smh,
                             float* __restrict__ negsum,
                             float* __restrict__ smsum) {
    int b = blockIdx.x / Qn;
    int q = blockIdx.x % Qn;
    const float* mask = pmasks + ((size_t)(b * Qn) + q) * (Hn * Wn);
    const float* pc = pts + (size_t)b * Pn * 2;
    float ns = 0.f, ss = 0.f;
    for (int p = threadIdx.x; p < Pn; p += blockDim.x) {
        float x = pc[2 * p + 0] * Wn - 0.5f;
        float y = pc[2 * p + 1] * Hn - 0.5f;
        float x0f = floorf(x), y0f = floorf(y);
        float wx = x - x0f, wy = y - y0f;
        int x0 = (int)x0f, y0 = (int)y0f;
        float v00 = samp2d(mask, y0, x0);
        float v01 = samp2d(mask, y0, x0 + 1);
        float v10 = samp2d(mask, y0 + 1, x0);
        float v11 = samp2d(mask, y0 + 1, x0 + 1);
        float pm = v00 * (1.f - wy) * (1.f - wx) + v01 * (1.f - wy) * wx
                 + v10 * wy * (1.f - wx) + v11 * wy * wx;
        float sm = sigmoid_f(pm);
        size_t idx = ((size_t)(b * Qn) + q) * Pn + p;
        pmh[idx] = (_Float16)pm;
        smh[idx] = (_Float16)sm;
        ns += softplus_f(pm);   // rowsum(neg_ce)
        ss += sm;               // rowsum(sigmoid) for dice denom
    }
    __shared__ float r1[256];
    __shared__ float r2[256];
    r1[threadIdx.x] = ns;
    r2[threadIdx.x] = ss;
    __syncthreads();
    for (int o = blockDim.x / 2; o > 0; o >>= 1) {
        if (threadIdx.x < (unsigned)o) {
            r1[threadIdx.x] += r1[threadIdx.x + o];
            r2[threadIdx.x] += r2[threadIdx.x + o];
        }
        __syncthreads();
    }
    if (threadIdx.x == 0) {
        negsum[b * Qn + q] = r1[0];
        smsum[b * Qn + q] = r2[0];
    }
}

// ---------------- Kernel 3: WMMA GEMM tiles + fused cost finalization
__global__ void hm_wmma_cost(const _Float16* __restrict__ pmh,
                             const _Float16* __restrict__ smh,
                             const _Float16* __restrict__ tmh,
                             const float* __restrict__ negsum,
                             const float* __restrict__ smsum,
                             const float* __restrict__ tmsum,
                             const float* __restrict__ logits,
                             const int* __restrict__ labels,
                             const float* __restrict__ pboxes,
                             const float* __restrict__ tboxes,
                             float* __restrict__ out) {
    const int tiles_per_b = MT * NT;
    int wave = threadIdx.x >> 5;
    int lane = threadIdx.x & 31;
    int tile = blockIdx.x * (blockDim.x >> 5) + wave;
    if (tile >= BS * tiles_per_b) return;   // wave-uniform: EXEC stays all-ones

    int b = tile / tiles_per_b;
    int rt = tile % tiles_per_b;
    int mtile = rt / NT;
    int ntile = rt % NT;
    int m0 = mtile * 16, n0 = ntile * 16;

    // clamp OOB rows/cols to valid data; those outputs are never stored
    int Ml = m0 + (lane & 15); if (Ml >= Qn) Ml = Qn - 1;
    int Nl = n0 + (lane & 15); if (Nl >= Tn) Nl = Tn - 1;

    const _Float16* pmrow = pmh + ((size_t)(b * Qn) + Ml) * Pn;
    const _Float16* smrow = smh + ((size_t)(b * Qn) + Ml) * Pn;
    const _Float16* tmrow = tmh + ((size_t)(b * Tn) + Nl) * Pn;
    int ksel = (lane >> 4) * 8;    // A: lane halves 16-31 take K+8 groups
    int bko  = (lane >> 4) * 16;   // unused: B loads use ksel, same K order as A

    v8f acc1 = {};   // pm @ tm^T
    v8f acc2 = {};   // sigmoid(pm) @ tm^T
    for (int kb = 0; kb < Pn; kb += 32) {
        v8h a0 = *(const v8h*)(pmrow + kb + ksel);        // K = kb+ksel .. +7
        v8h a1 = *(const v8h*)(pmrow + kb + 16 + ksel);   // K = kb+16+ksel .. +7
        v8h s0 = *(const v8h*)(smrow + kb + ksel);
        v8h s1 = *(const v8h*)(smrow + kb + 16 + ksel);
        v8h b0 = *(const v8h*)(tmrow + kb + ksel);        // K = kb+ksel .. +7
        v8h b1 = *(const v8h*)(tmrow + kb + 16 + ksel);   // K = kb+16+ksel .. +7
        v16h bfrag;
#pragma unroll
        for (int i = 0; i < 8; i++) { bfrag[i] = b0[i]; bfrag[i + 8] = b1[i]; }
        v16h a_pm, a_sm;
#pragma unroll
        for (int i = 0; i < 8; i++) {
            a_pm[i] = a0[i]; a_pm[i + 8] = a1[i];
            a_sm[i] = s0[i]; a_sm[i + 8] = s1[i];
        }
        acc1 = __builtin_amdgcn_wmma_f32_16x16x32_f16(
            false, a_pm, false, bfrag, (short)0, acc1, false, false);
        acc2 = __builtin_amdgcn_wmma_f32_16x16x32_f16(
            false, a_sm, false, bfrag, (short)0, acc2, false, false);
    }

    // ---- finalize: lane owns column t = n0 + lane%16, rows M = r + 8*(lane/16)
    int t = n0 + (lane & 15);
    bool tvalid = (t < Tn);
    int tt = tvalid ? t : (Tn - 1);
    int lab = labels[b * Tn + tt];
    float tsum = tmsum[b * Tn + tt];
    const float* tb = tboxes + ((size_t)(b * Tn) + tt) * 4;
    float tcx = tb[0], tcy = tb[1], tw = tb[2], th = tb[3];
    float tx0 = tcx - 0.5f * tw, ty0 = tcy - 0.5f * th;
    float tx1 = tcx + 0.5f * tw, ty1 = tcy + 0.5f * th;
    float ta = (tx1 - tx0) * (ty1 - ty0);

#pragma unroll
    for (int r = 0; r < 8; r++) {
        int q = m0 + r + 8 * (lane >> 4);
        if (q >= Qn || !tvalid) continue;
        const float* pb = pboxes + ((size_t)(b * Qn) + q) * 4;
        float pcx = pb[0], pcy = pb[1], pw = pb[2], ph = pb[3];
        float cost_bbox = fabsf(pcx - tcx) + fabsf(pcy - tcy)
                        + fabsf(pw - tw) + fabsf(ph - th);
        float px0 = pcx - 0.5f * pw, py0 = pcy - 0.5f * ph;
        float px1 = pcx + 0.5f * pw, py1 = pcy + 0.5f * ph;
        float pa = (px1 - px0) * (py1 - py0);
        float iw = fmaxf(fminf(px1, tx1) - fmaxf(px0, tx0), 0.f);
        float ih = fmaxf(fminf(py1, ty1) - fmaxf(py0, ty0), 0.f);
        float inter = iw * ih;
        float uni = pa + ta - inter;
        float iou = inter / uni;
        float cw = fmaxf(fmaxf(px1, tx1) - fminf(px0, tx0), 0.f);
        float chh = fmaxf(fmaxf(py1, ty1) - fminf(py0, ty0), 0.f);
        float carea = cw * chh;
        float giou = iou - (carea - uni) / carea;

        float L = logits[((size_t)(b * Qn) + q) * Cn + lab];
        float pr = sigmoid_f(L);
        float pos = 0.25f * (1.f - pr) * (1.f - pr) * softplus_f(-L);
        float neg = 0.75f * pr * pr * softplus_f(L);
        float cclass = pos - neg;

        float S1 = acc1[r];
        float S2 = acc2[r];
        // ce = (pos_ce@tm^T + neg_ce@(1-tm)^T)/P = (negsum - pm@tm^T)/P
        float ce = (negsum[b * Qn + q] - S1) * (1.f / (float)Pn);
        float dice = 1.f - (2.f * S2 + 1.f) / (smsum[b * Qn + q] + tsum + 1.f);

        out[((size_t)(b * Qn) + q) * Tn + t] =
            cost_bbox + cclass - giou + ce + dice;
    }
}

// ---------------- Kernel 4: per-batch max over finite entries
__global__ void hm_reduce_max(const float* __restrict__ out, float* __restrict__ mx) {
    int b = blockIdx.x;
    float m = -INFINITY;
    for (int i = threadIdx.x; i < Qn * Tn; i += blockDim.x) {
        float v = out[(size_t)b * Qn * Tn + i];
        if (__builtin_isfinite(v)) m = fmaxf(m, v);
    }
    __shared__ float red[256];
    red[threadIdx.x] = m;
    __syncthreads();
    for (int o = blockDim.x / 2; o > 0; o >>= 1) {
        if (threadIdx.x < (unsigned)o)
            red[threadIdx.x] = fmaxf(red[threadIdx.x], red[threadIdx.x + o]);
        __syncthreads();
    }
    if (threadIdx.x == 0) mx[b] = red[0];
}

// ---------------- Kernel 5: replace non-finite with 2*mx
__global__ void hm_patch(float* __restrict__ out, const float* __restrict__ mx) {
    size_t i = (size_t)blockIdx.x * blockDim.x + threadIdx.x;
    if (i >= (size_t)BS * Qn * Tn) return;
    int b = (int)(i / (size_t)(Qn * Tn));
    float v = out[i];
    if (!__builtin_isfinite(v)) out[i] = 2.f * mx[b];
}

extern "C" void kernel_launch(void* const* d_in, const int* in_sizes, int n_in,
                              void* d_out, int out_size, void* d_ws, size_t ws_size,
                              hipStream_t stream) {
    (void)in_sizes; (void)n_in; (void)out_size; (void)ws_size;
    const float* logits = (const float*)d_in[0];
    const float* pboxes = (const float*)d_in[1];
    const int*   labels = (const int*)d_in[2];
    const float* tboxes = (const float*)d_in[3];
    const float* pmasks = (const float*)d_in[4];
    const int*   tmasks = (const int*)d_in[5];
    const float* pts    = (const float*)d_in[6];
    float* out = (float*)d_out;

    // carve workspace (256B-aligned slices); total ~7.8 MB
    char* ws = (char*)d_ws;
    size_t off = 0;
    auto carve = [&](size_t bytes) -> char* {
        char* p = ws + off;
        off = (off + bytes + 255) & ~(size_t)255;
        return p;
    };
    _Float16* pmh    = (_Float16*)carve((size_t)BS * Qn * Pn * sizeof(_Float16));
    _Float16* smh    = (_Float16*)carve((size_t)BS * Qn * Pn * sizeof(_Float16));
    _Float16* tmh    = (_Float16*)carve((size_t)BS * Tn * Pn * sizeof(_Float16));
    float*    tmsum  = (float*)carve((size_t)BS * Tn * sizeof(float));
    float*    negsum = (float*)carve((size_t)BS * Qn * sizeof(float));
    float*    smsum  = (float*)carve((size_t)BS * Qn * sizeof(float));
    float*    mx     = (float*)carve((size_t)BS * sizeof(float));

    hm_tm_sample<<<BS * Tn, 256, 0, stream>>>(tmasks, pts, tmh, tmsum);
    hm_pm_sample<<<BS * Qn, 256, 0, stream>>>(pmasks, pts, pmh, smh, negsum, smsum);

    int total_tiles = BS * MT * NT;          // 798 waves
    int waves_per_block = 4;                 // 128 threads = 4 wave32
    int blocks = (total_tiles + waves_per_block - 1) / waves_per_block;
    hm_wmma_cost<<<blocks, 128, 0, stream>>>(pmh, smh, tmh, negsum, smsum, tmsum,
                                             logits, labels, pboxes, tboxes, out);

    hm_reduce_max<<<BS, 256, 0, stream>>>(out, mx);
    int tot = BS * Qn * Tn;
    hm_patch<<<(tot + 255) / 256, 256, 0, stream>>>(out, mx);
}
